// Attention_52063593562622
// MI455X (gfx1250) — matrix operand order repaired and checked by reading, not compiled, not run on hardware
//
#include <hip/hip_runtime.h>

typedef __attribute__((ext_vector_type(16))) _Float16 v16h;
typedef __attribute__((ext_vector_type(8)))  _Float16 v8h;
typedef __attribute__((ext_vector_type(4)))  _Float16 v4h;
typedef __attribute__((ext_vector_type(8)))  float    v8f;

constexpr int Bc = 16, Hc = 4, Sc = 1024, Dc = 128;
constexpr size_t Nqkv = (size_t)Bc * Hc * Sc * Dc;   // 8,388,608 elements
constexpr int ROWPAD    = 4;
constexpr int ROWSTRIDE = Sc + ROWPAD;               // 1028 floats per score row
constexpr int WAVE_SC   = 16 * ROWSTRIDE;
constexpr int WAVE_LDS  = WAVE_SC + 32;              // + rmax[16] + rsum[16]
constexpr float SCALE   = 0.08838834764831845f;      // 1/sqrt(128)

// ---------------- preprocessing: f32 -> f16 convert (row-major) ----------------
__global__ __launch_bounds__(256) void cvt_f16_kernel(const float* __restrict__ src,
                                                      _Float16* __restrict__ dst) {
  const size_t i = (size_t)blockIdx.x * blockDim.x + threadIdx.x;  // 4 elems/thread
  float4 f = ((const float4*)src)[i];
  v4h h; h[0] = (_Float16)f.x; h[1] = (_Float16)f.y;
         h[2] = (_Float16)f.z; h[3] = (_Float16)f.w;
  ((v4h*)dst)[i] = h;
}

// ---------------- preprocessing: V -> Vt[d][s] f16 (per b,h) -------------------
__global__ __launch_bounds__(256) void vt_kernel(const float* __restrict__ V,
                                                 _Float16* __restrict__ Vt) {
  __shared__ float tile[32][33];
  int bid = blockIdx.x;
  const int dt = bid & 3;  bid >>= 2;     // D/32 = 4
  const int st = bid & 31; bid >>= 5;     // S/32 = 32
  const int bh = bid;                     // B*H = 64
  const int tx = threadIdx.x & 31, ty = threadIdx.x >> 5;

  const float* vb = V + ((size_t)bh * Sc + st * 32) * Dc + dt * 32;
  #pragma unroll
  for (int i = 0; i < 4; ++i) {
    const int srow = ty + 8 * i;
    tile[srow][tx] = vb[srow * Dc + tx];            // coalesced along d
  }
  __syncthreads();
  _Float16* ob = Vt + ((size_t)bh * Dc + dt * 32) * Sc + st * 32;
  #pragma unroll
  for (int i = 0; i < 4; ++i) {
    const int drow = ty + 8 * i;
    ob[drow * Sc + tx] = (_Float16)tile[tx][drow];  // coalesced along s
  }
}

// ---------------- fused attention ----------------------------------------------
template <bool F16>
__global__ __launch_bounds__(128) void attn_fused_kernel(
    const float* __restrict__ Q, const float* __restrict__ K,
    const float* __restrict__ V, const float* __restrict__ T,
    const int*   __restrict__ M,
    const _Float16* __restrict__ Qh, const _Float16* __restrict__ Kh,
    const _Float16* __restrict__ Vt,
    float* __restrict__ Out, float* __restrict__ P)
{
  extern __shared__ float lds[];

  const int lane = threadIdx.x & 31;
  const int wave = threadIdx.x >> 5;       // = head index
  const int h    = wave;
  const int b    = blockIdx.x >> 6;        // S/16 = 64 q-tiles per b
  const int q0   = (blockIdx.x & 63) * 16;
  const int bh   = b * Hc + h;

  float* sc    = lds + wave * WAVE_LDS;
  float* rmaxL = sc + WAVE_SC;
  float* rsumL = rmaxL + 16;

  const int half  = lane >> 4;
  const int l15   = lane & 15;
  const int abase = 8 * half;              // A-layout k sub-offset
  const int kb16  = 8 * half;              // B-layout k sub-offset

  // ---- Q tile into WMMA A-layout (D=128 -> 4 chunks of k32)
  v16h aq[4];
  if constexpr (F16) {
    const _Float16* qrow = Qh + (((size_t)bh * Sc) + q0 + l15) * Dc;
    #pragma unroll
    for (int c = 0; c < 4; ++c) {
      v8h lo = *(const v8h*)(qrow + 32 * c + abase);
      v8h hi = *(const v8h*)(qrow + 32 * c + 16 + abase);
      aq[c] = __builtin_shufflevector(lo, hi, 0,1,2,3,4,5,6,7,8,9,10,11,12,13,14,15);
    }
  } else {
    const float* qrow = Q + (((size_t)bh * Sc) + q0 + l15) * Dc;
    #pragma unroll
    for (int c = 0; c < 4; ++c) {
      const float* p0 = qrow + 32 * c + abase;
      const float* p1 = qrow + 32 * c + 16 + abase;
      #pragma unroll
      for (int i = 0; i < 8; ++i) {
        aq[c][i]     = (_Float16)p0[i];
        aq[c][8 + i] = (_Float16)p1[i];
      }
    }
  }

  // ---- pass 1: scores = (Q K^T)*scale*time, mask, row-max, -> LDS
  const float*    kbp   = K  + ((size_t)bh * Sc) * Dc;
  const _Float16* kbph  = Kh + ((size_t)bh * Sc) * Dc;
  const float*    trow0 = T + ((size_t)b * Sc + q0) * Sc;
  const int*      mrow0 = M + ((size_t)b * Sc + q0) * Sc;

  // loop-invariant 32-bit lane offsets (-> saddr + voffset addressing)
  const int klaneoff = l15 * Dc + kb16;    // for K rows
  const int vlaneoff = l15 * Sc + kb16;    // for Vt rows

  float pm[8];
  #pragma unroll
  for (int r = 0; r < 8; ++r) pm[r] = -3.0e38f;

  const v8f zero8 = {0.f,0.f,0.f,0.f,0.f,0.f,0.f,0.f};

  for (int t = 0; t < 64; ++t) {
    v16h bk[4];
    if constexpr (F16) {
      const _Float16* krow_t = kbph + t * (16 * Dc);    // uniform base bump
      if (t + 1 < 64) __builtin_prefetch(krow_t + 16 * Dc + klaneoff, 0, 0);
      #pragma unroll
      for (int c = 0; c < 4; ++c) {
        v8h lo = *(const v8h*)(krow_t + klaneoff + 32 * c);
        v8h hi = *(const v8h*)(krow_t + klaneoff + 32 * c + 16);
        bk[c] = __builtin_shufflevector(lo, hi, 0,1,2,3,4,5,6,7,8,9,10,11,12,13,14,15);
      }
    } else {
      const float* krow_t = kbp + t * (16 * Dc);
      if (t + 1 < 64) __builtin_prefetch(krow_t + 16 * Dc + klaneoff, 0, 0);
      #pragma unroll
      for (int c = 0; c < 4; ++c) {
        const float* pk = krow_t + klaneoff + 32 * c;
        #pragma unroll
        for (int i = 0; i < 16; ++i) bk[c][i] = (_Float16)pk[i + (i & 8)];
      }
    }

    v8f acc = zero8;
    #pragma unroll
    for (int c = 0; c < 4; ++c)
      acc = __builtin_amdgcn_wmma_f32_16x16x32_f16(
          false, aq[c], false, bk[c], (short)0, acc, false, false);

    const float* tq = trow0 + 16 * t;      // uniform base bumps
    const int*   mq = mrow0 + 16 * t;
    const int    n  = t * 16 + l15;        // C-layout column per lane
    #pragma unroll
    for (int r = 0; r < 8; ++r) {
      const int m   = r + 8 * half;        // C-layout row
      const int off = m * Sc + l15;        // 32-bit lane offset
      float s = acc[r] * SCALE * tq[off];
      const int mk = mq[off];
      s = (mk == 0) ? -1.0e9f : s;
      pm[r] = fmaxf(pm[r], s);
      sc[m * ROWSTRIDE + n] = s;
    }
  }

  // row-max reduction across the 16 lanes of each half
  #pragma unroll
  for (int r = 0; r < 8; ++r) {
    float m0 = pm[r];
    m0 = fmaxf(m0, __shfl_xor(m0, 1, 32));
    m0 = fmaxf(m0, __shfl_xor(m0, 2, 32));
    m0 = fmaxf(m0, __shfl_xor(m0, 4, 32));
    m0 = fmaxf(m0, __shfl_xor(m0, 8, 32));
    pm[r] = m0;
  }
  #pragma unroll
  for (int r = 0; r < 8; ++r)
    if (l15 == r) rmaxL[half * 8 + r] = pm[r];

  // ---- pass 2: e = exp(s-max) (A-layout), row-sum, O += P V
  v8f oacc[8];
  #pragma unroll
  for (int j = 0; j < 8; ++j) oacc[j] = zero8;

  const float*    vbp = V  + ((size_t)bh * Sc) * Dc;
  const _Float16* vtb = Vt + (size_t)bh * Dc * Sc;   // Vt[d][s]
  const float rmax_m = rmaxL[l15];
  float psum = 0.f;

  for (int u = 0; u < 32; ++u) {
    float* srow = sc + l15 * ROWSTRIDE + 32 * u + abase;
    // vectorized LDS read: 4x ds_load_b128
    float4 a0 = *(const float4*)(srow + 0);
    float4 a1 = *(const float4*)(srow + 4);
    float4 b0 = *(const float4*)(srow + 16);
    float4 b1 = *(const float4*)(srow + 20);
    float sv[16] = {a0.x,a0.y,a0.z,a0.w, a1.x,a1.y,a1.z,a1.w,
                    b0.x,b0.y,b0.z,b0.w, b1.x,b1.y,b1.z,b1.w};
    float ev[16];
    v16h ea;
    #pragma unroll
    for (int i = 0; i < 16; ++i) {
      ev[i] = __expf(sv[i] - rmax_m);
      psum += ev[i];
      ea[i] = (_Float16)ev[i];
    }
    // write e back (for p_attn pass): 4x ds_store_b128
    *(float4*)(srow + 0)  = make_float4(ev[0],  ev[1],  ev[2],  ev[3]);
    *(float4*)(srow + 4)  = make_float4(ev[4],  ev[5],  ev[6],  ev[7]);
    *(float4*)(srow + 16) = make_float4(ev[8],  ev[9],  ev[10], ev[11]);
    *(float4*)(srow + 20) = make_float4(ev[12], ev[13], ev[14], ev[15]);

    const _Float16* vtu = vtb + 32 * u;    // uniform base bump
    #pragma unroll
    for (int j = 0; j < 8; ++j) {
      v16h bv;
      if constexpr (F16) {
        // Vt row d = 16j+l15: lane-const voffset, j folded into imm offset
        const _Float16* vp = vtu + vlaneoff + j * (16 * Sc);
        v8h lo = *(const v8h*)(vp);
        v8h hi = *(const v8h*)(vp + 16);
        bv = __builtin_shufflevector(lo, hi, 0,1,2,3,4,5,6,7,8,9,10,11,12,13,14,15);
      } else {
        const float* vp = vbp + (32 * u + kb16) * Dc + 16 * j + l15;
        #pragma unroll
        for (int i = 0; i < 16; ++i) bv[i] = (_Float16)vp[(i + (i & 8)) * Dc];
      }
      oacc[j] = __builtin_amdgcn_wmma_f32_16x16x32_f16(
          false, ea, false, bv, (short)0, oacc[j], false, false);
    }
  }

  psum += __shfl_xor(psum, 16, 32);
  if (lane < 16) rsumL[lane] = psum;

  // ---- pass 3: normalize + store O and p_attn
  float* ob = Out + (((size_t)bh * Sc) + q0) * Dc;
  #pragma unroll
  for (int r = 0; r < 8; ++r) {
    const int m = r + 8 * half;
    const float inv = 1.0f / fmaxf(rsumL[m], 1e-30f);
    #pragma unroll
    for (int j = 0; j < 8; ++j)
      ob[m * Dc + 16 * j + l15] = oacc[j][r] * inv;
  }

  float* pb = P + (((size_t)bh * Sc) + q0) * Sc;
  for (int m = 0; m < 16; ++m) {
    const float inv = 1.0f / fmaxf(rsumL[m], 1e-30f);
    const float* er = sc + m * ROWSTRIDE;
    float* pr = pb + m * Sc;
    #pragma unroll
    for (int i = 0; i < 8; ++i) {
      const int n = i * 128 + 4 * lane;
      float4 e4 = *(const float4*)(er + n);
      float4 o4 = make_float4(e4.x * inv, e4.y * inv, e4.z * inv, e4.w * inv);
      *(float4*)(pr + n) = o4;
    }
  }
}

extern "C" void kernel_launch(void* const* d_in, const int* in_sizes, int n_in,
                              void* d_out, int out_size, void* d_ws, size_t ws_size,
                              hipStream_t stream) {
  const float* q = (const float*)d_in[0];
  const float* k = (const float*)d_in[1];
  const float* v = (const float*)d_in[2];
  const float* t = (const float*)d_in[3];
  const int*   m = (const int*)d_in[4];

  float* out = (float*)d_out;                        // [B,H,S,D]
  float* p   = out + Nqkv;                           // [B,H,S,S]

  const size_t shmem = (size_t)4 * WAVE_LDS * sizeof(float);  // ~258 KB / WGP
  dim3 grid(Bc * (Sc / 16));   // 1024 workgroups
  dim3 block(128);             // 4 waves: one head each; shared time/mask in L2

  const size_t ws_need = 3 * Nqkv * sizeof(_Float16);         // Qh, Kh, Vt

  if (ws_size >= ws_need) {
    _Float16* qh = (_Float16*)d_ws;
    _Float16* kh = qh + Nqkv;
    _Float16* vt = kh + Nqkv;

    const int nblk = (int)(Nqkv / 4 / 256);          // 4 elems/thread
    cvt_f16_kernel<<<nblk, 256, 0, stream>>>(q, qh);
    cvt_f16_kernel<<<nblk, 256, 0, stream>>>(k, kh);
    vt_kernel<<<Bc * Hc * (Sc / 32) * (Dc / 32), 256, 0, stream>>>(v, vt);

    (void)hipFuncSetAttribute((const void*)attn_fused_kernel<true>,
                              hipFuncAttributeMaxDynamicSharedMemorySize, (int)shmem);
    attn_fused_kernel<true><<<grid, block, shmem, stream>>>(
        q, k, v, t, m, qh, kh, vt, out, p);
  } else {
    (void)hipFuncSetAttribute((const void*)attn_fused_kernel<false>,
                              hipFuncAttributeMaxDynamicSharedMemorySize, (int)shmem);
    attn_fused_kernel<false><<<grid, block, shmem, stream>>>(
        q, k, v, t, m, nullptr, nullptr, nullptr, out, p);
  }
}
